// MultiHeadAttention_88115549045474
// MI455X (gfx1250) — compile-verified, matrix operand order checked by reading, not run on hardware
//
#include <hip/hip_runtime.h>
#include <hip/hip_bf16.h>

#define B_  2
#define S_  2048
#define D_  1024
#define H_  16
#define HD_ 64

typedef __attribute__((ext_vector_type(16))) __bf16 v16bf;
typedef __attribute__((ext_vector_type(8)))  __bf16 v8bf;
typedef __attribute__((ext_vector_type(2)))  __bf16 v2bf;
typedef __attribute__((ext_vector_type(8)))  float  v8f;
typedef __attribute__((ext_vector_type(2)))  float  v2f;

union FragBF { v16bf v; v8bf h[2]; };

// Per-lane async DMA: 16B global -> LDS, tracked by ASYNCcnt.
#define ASYNC_LOAD_B128(ldsaddr, gaddr)                                        \
  asm volatile("global_load_async_to_lds_b128 %0, %1, off"                     \
               ::"v"(ldsaddr), "v"(gaddr) : "memory")
#define WAIT_ASYNC(n) asm volatile("s_wait_asynccnt " #n ::: "memory")
// Compiler-only scheduling fence: keep fragment loads grouped ahead of WMMAs
// so one (partial) s_wait_dscnt chain covers a whole WMMA block.
#define SCHED_FENCE() asm volatile("" ::: "memory")

// Native bf16 convert (gfx1250 has HW BF16 converts, RNE).
static __device__ __forceinline__ __bf16 f2bf(float f) { return (__bf16)f; }

// 16x32 bf16 A-fragment (or B^T row-major fragment), wave32 WMMA layout:
// lane l(0-15): row=rowbase+l, k-chunks {kbase, kbase+16}
// lane l(16-31): row=rowbase+l-16, k-chunks {kbase+8, kbase+24}
static __device__ __forceinline__ v16bf load_frag(const __bf16* base, int row, int ld,
                                                  int kbase, int lane) {
  int half = (lane >> 4) & 1;
  const __bf16* p = base + row * ld + kbase + half * 8;
  FragBF f;
  f.h[0] = *(const v8bf*)(p);
  f.h[1] = *(const v8bf*)(p + 16);
  return f.v;
}

static __device__ __forceinline__ v8f wmma_bf16(v16bf a, v16bf b, v8f c) {
  return __builtin_amdgcn_wmma_f32_16x16x32_bf16(false, a, false, b, (short)0, c,
                                                 false, false);
}

// ---------------------------------------------------------------- fp32 -> bf16
// Pairwise so clang can fuse into packed HW converts + 4B stores.
__global__ void cvt_bf16(const float* __restrict__ in, __bf16* __restrict__ out,
                         int n) {
  int stride = gridDim.x * blockDim.x;
  for (int p = blockIdx.x * blockDim.x + threadIdx.x; 2 * p < n; p += stride) {
    v2f x = *(const v2f*)(in + 2 * p);
    v2bf y;
    y.x = (__bf16)x.x;
    y.y = (__bf16)x.y;
    *(v2bf*)(out + 2 * p) = y;
  }
}

// ------------------------------------------------- QKV GEMM: [4096,1024]x[1024,3072]
// Block tile 256(M) x 64(N): 8 waves own 32 rows each; W tile (64x32 bf16, 4KB)
// double-buffered in LDS via async DMA, shared by all 8 waves. 8 WMMA per wave
// per k-step (2 A-frags x 4 W-frags). Scatter: Q,K -> [B,H,S,64], V -> [B,H,64,S].
__global__ __launch_bounds__(256) void qkv_gemm(
    const __bf16* __restrict__ X, const __bf16* __restrict__ W,
    const float* __restrict__ bias,
    __bf16* __restrict__ Qb, __bf16* __restrict__ Kb, __bf16* __restrict__ Vt) {
  __shared__ __align__(16) __bf16 Wlds[2][64 * 32];     // 2 x 4KB
  int lane = threadIdx.x & 31;
  int wv   = threadIdx.x >> 5;
  int lm = lane & 15, half = lane >> 4;
  int n0 = blockIdx.x * 64;
  int m0 = blockIdx.y * 256 + wv * 32;

  auto issue_w = [&](int k, int buf) {
    int c = threadIdx.x;                 // 256 chunks of 16B = 4KB tile
    int row = c >> 2, col = (c & 3) << 3;
    unsigned long long g =
        (unsigned long long)(size_t)(W + (n0 + row) * D_ + k + col);
    unsigned int l = (unsigned int)(size_t)&Wlds[buf][row * 32 + col];
    ASYNC_LOAD_B128(l, g);
  };

  issue_w(0, 0);
  v8f acc[2][4];
#pragma unroll
  for (int i = 0; i < 2; ++i)
#pragma unroll
    for (int j = 0; j < 4; ++j) acc[i][j] = v8f{};

  for (int k = 0; k < D_; k += 32) {
    int buf = (k >> 5) & 1;
    if (k + 32 < D_) { issue_w(k + 32, buf ^ 1); WAIT_ASYNC(1); }
    else             { WAIT_ASYNC(0); }
    __syncthreads();
    v16bf a0 = load_frag(X, m0 + lm,      D_, k, lane);
    v16bf a1 = load_frag(X, m0 + 16 + lm, D_, k, lane);
    v16bf w0 = load_frag(&Wlds[buf][0],  0 + lm, 32, 0, lane);
    v16bf w1 = load_frag(&Wlds[buf][0], 16 + lm, 32, 0, lane);
    v16bf w2 = load_frag(&Wlds[buf][0], 32 + lm, 32, 0, lane);
    v16bf w3 = load_frag(&Wlds[buf][0], 48 + lm, 32, 0, lane);
    SCHED_FENCE();
    acc[0][0] = wmma_bf16(a0, w0, acc[0][0]);
    acc[0][1] = wmma_bf16(a0, w1, acc[0][1]);
    acc[0][2] = wmma_bf16(a0, w2, acc[0][2]);
    acc[0][3] = wmma_bf16(a0, w3, acc[0][3]);
    acc[1][0] = wmma_bf16(a1, w0, acc[1][0]);
    acc[1][1] = wmma_bf16(a1, w1, acc[1][1]);
    acc[1][2] = wmma_bf16(a1, w2, acc[1][2]);
    acc[1][3] = wmma_bf16(a1, w3, acc[1][3]);
    __syncthreads();
  }

#pragma unroll
  for (int i = 0; i < 2; ++i) {
#pragma unroll
    for (int j = 0; j < 4; ++j) {
#pragma unroll
      for (int r = 0; r < 8; ++r) {
        int m = m0 + i * 16 + r + half * 8;         // row in [B*S]
        int n = n0 + j * 16 + lm;                   // column in [3D]
        float v = acc[i][j][r] + bias[n];
        int bb = m / S_, s = m % S_;
        int h = n / (3 * HD_), jj = n % (3 * HD_);
        if (jj < HD_)
          Qb[((bb * H_ + h) * S_ + s) * HD_ + jj] = f2bf(v);
        else if (jj < 2 * HD_)
          Kb[((bb * H_ + h) * S_ + s) * HD_ + (jj - HD_)] = f2bf(v);
        else
          Vt[((bb * H_ + h) * HD_ + (jj - 2 * HD_)) * S_ + s] = f2bf(v);
      }
    }
  }
}

// ------------------------------------------------------------ flash attention
// grid = (B*H, S/64); 128 threads = 4 waves; 16 query rows per wave; KV tile = 64.
// K (64x64) and V^T (64x64) tiles double-buffered in LDS via async DMA (shared
// by all 4 waves), pipelined with s_wait_asynccnt. 16 WMMA per iteration.
__global__ __launch_bounds__(128) void attn(
    const __bf16* __restrict__ Qb, const __bf16* __restrict__ Kb,
    const __bf16* __restrict__ Vt, const float* __restrict__ mask,
    __bf16* __restrict__ vals) {
  __shared__ __align__(16) __bf16 Kbuf[2][64 * 64];     // 2 x 8KB
  __shared__ __align__(16) __bf16 Vbuf[2][64 * 64];     // 2 x 8KB
  __shared__ __align__(16) __bf16 plds[4][16][72];      // P transpose bounce
  int lane = threadIdx.x & 31;
  int wv   = threadIdx.x >> 5;
  int lm = lane & 15, half = lane >> 4;
  int bh = blockIdx.x;
  int b = bh / H_, h = bh % H_;
  int q0 = blockIdx.y * 64 + wv * 16;

  const __bf16* Qp = Qb + bh * S_ * HD_;
  const __bf16* Kp = Kb + bh * S_ * HD_;
  const __bf16* Vp = Vt + bh * HD_ * S_;

  auto issue_kv = [&](int kt, int buf) {
    int t = threadIdx.x;
#pragma unroll
    for (int c = t; c < 512; c += 128) {    // K tile: 64 rows x 64 bf16 (8KB)
      int row = c >> 3, col = (c & 7) << 3;
      unsigned long long g =
          (unsigned long long)(size_t)(Kp + (kt + row) * HD_ + col);
      unsigned int l = (unsigned int)(size_t)&Kbuf[buf][row * 64 + col];
      ASYNC_LOAD_B128(l, g);
    }
#pragma unroll
    for (int c = t; c < 512; c += 128) {    // V^T tile: 64 hd rows x 64 s (8KB)
      int row = c >> 3, col = (c & 7) << 3;
      unsigned long long g =
          (unsigned long long)(size_t)(Vp + row * S_ + kt + col);
      unsigned int l = (unsigned int)(size_t)&Vbuf[buf][row * 64 + col];
      ASYNC_LOAD_B128(l, g);
    }
  };

  v16bf aq0 = load_frag(Qp, q0 + lm, HD_, 0,  lane);
  v16bf aq1 = load_frag(Qp, q0 + lm, HD_, 32, lane);

  v8f o[4];
#pragma unroll
  for (int j = 0; j < 4; ++j) o[j] = v8f{};
  float mrow[8], lrow[8];
#pragma unroll
  for (int r = 0; r < 8; ++r) { mrow[r] = -1e30f; lrow[r] = 0.f; }

  issue_kv(0, 0);
  for (int kt = 0; kt < S_; kt += 64) {
    int buf = (kt >> 6) & 1;
    if (kt + 64 < S_) { issue_kv(kt + 64, buf ^ 1); WAIT_ASYNC(8); }
    else              { WAIT_ASYNC(0); }
    __syncthreads();

    // ---- S tile = Q[16,64] @ K^T[64,64] from LDS-staged K tile
    v8f s[4];
#pragma unroll
    for (int j = 0; j < 4; j += 2) {
      v16bf b00 = load_frag(&Kbuf[buf][0], (j + 0) * 16 + lm, 64, 0,  lane);
      v16bf b01 = load_frag(&Kbuf[buf][0], (j + 0) * 16 + lm, 64, 32, lane);
      v16bf b10 = load_frag(&Kbuf[buf][0], (j + 1) * 16 + lm, 64, 0,  lane);
      v16bf b11 = load_frag(&Kbuf[buf][0], (j + 1) * 16 + lm, 64, 32, lane);
      SCHED_FENCE();
      v8f t0 = {}, t1 = {};
      t0 = wmma_bf16(aq0, b00, t0);
      t0 = wmma_bf16(aq1, b01, t0);
      t1 = wmma_bf16(aq0, b10, t1);
      t1 = wmma_bf16(aq1, b11, t1);
      s[j] = t0; s[j + 1] = t1;
    }

    const float scale = 0.125f;                 // 1/sqrt(64)
#pragma unroll
    for (int j = 0; j < 4; ++j)
#pragma unroll
      for (int r = 0; r < 8; ++r) {
        int qr = q0 + r + half * 8;
        s[j][r] = s[j][r] * scale + mask[(b * S_ + qr) * S_ + kt + j * 16 + lm];
      }

    // ---- online softmax (rows split across lane halves; reduce within 16 lanes)
#pragma unroll
    for (int r = 0; r < 8; ++r) {
      float mx = fmaxf(fmaxf(s[0][r], s[1][r]), fmaxf(s[2][r], s[3][r]));
      mx = fmaxf(mx, __shfl_xor(mx, 1, 32));
      mx = fmaxf(mx, __shfl_xor(mx, 2, 32));
      mx = fmaxf(mx, __shfl_xor(mx, 4, 32));
      mx = fmaxf(mx, __shfl_xor(mx, 8, 32));
      float mn = fmaxf(mrow[r], mx);
      float alpha = __expf(mrow[r] - mn);
      mrow[r] = mn;
      float ps = 0.f;
#pragma unroll
      for (int j = 0; j < 4; ++j) {
        float p = __expf(s[j][r] - mn);
        s[j][r] = p;
        ps += p;
      }
      ps += __shfl_xor(ps, 1, 32);
      ps += __shfl_xor(ps, 2, 32);
      ps += __shfl_xor(ps, 4, 32);
      ps += __shfl_xor(ps, 8, 32);
      lrow[r] = lrow[r] * alpha + ps;
#pragma unroll
      for (int j = 0; j < 4; ++j) o[j][r] *= alpha;
    }

    // ---- P: C-layout -> A-layout via per-wave LDS (same-wave, dscnt wait only)
#pragma unroll
    for (int j = 0; j < 4; ++j)
#pragma unroll
      for (int r = 0; r < 8; ++r)
        plds[wv][r + half * 8][j * 16 + lm] = f2bf(s[j][r]);
    asm volatile("s_wait_dscnt 0" ::: "memory");
    FragBF pa0, pa1;
    pa0.h[0] = *(const v8bf*)&plds[wv][lm][half * 8];
    pa0.h[1] = *(const v8bf*)&plds[wv][lm][16 + half * 8];
    pa1.h[0] = *(const v8bf*)&plds[wv][lm][32 + half * 8];
    pa1.h[1] = *(const v8bf*)&plds[wv][lm][48 + half * 8];

    // ---- O += P[16,64] @ V[64,64] from LDS-staged V^T tile
#pragma unroll
    for (int j = 0; j < 4; j += 2) {
      v16bf v00 = load_frag(&Vbuf[buf][0], (j + 0) * 16 + lm, 64, 0,  lane);
      v16bf v01 = load_frag(&Vbuf[buf][0], (j + 0) * 16 + lm, 64, 32, lane);
      v16bf v10 = load_frag(&Vbuf[buf][0], (j + 1) * 16 + lm, 64, 0,  lane);
      v16bf v11 = load_frag(&Vbuf[buf][0], (j + 1) * 16 + lm, 64, 32, lane);
      SCHED_FENCE();
      o[j]     = wmma_bf16(pa0.v, v00, o[j]);
      o[j]     = wmma_bf16(pa1.v, v01, o[j]);
      o[j + 1] = wmma_bf16(pa0.v, v10, o[j + 1]);
      o[j + 1] = wmma_bf16(pa1.v, v11, o[j + 1]);
    }

    __syncthreads();                            // protect LDS tiles before reuse
  }

  // ---- normalize + store into vals[B,S,D] (heads interleaved) as bf16
#pragma unroll
  for (int r = 0; r < 8; ++r) {
    float inv = 1.0f / lrow[r];
    int srow = q0 + r + half * 8;
    __bf16* vp = vals + (b * S_ + srow) * D_ + h * HD_;
#pragma unroll
    for (int j = 0; j < 4; ++j)
      vp[j * 16 + lm] = f2bf(o[j][r] * inv);
  }
}

// ------------------------------------------------ out GEMM: [4096,1024]x[1024,1024]
__global__ __launch_bounds__(256) void out_gemm(
    const __bf16* __restrict__ A, const __bf16* __restrict__ W,
    const float* __restrict__ bias, float* __restrict__ out) {
  __shared__ __align__(16) __bf16 Wlds[2][64 * 32];
  int lane = threadIdx.x & 31;
  int wv   = threadIdx.x >> 5;
  int lm = lane & 15, half = lane >> 4;
  int n0 = blockIdx.x * 64;
  int m0 = blockIdx.y * 256 + wv * 32;

  auto issue_w = [&](int k, int buf) {
    int c = threadIdx.x;
    int row = c >> 2, col = (c & 3) << 3;
    unsigned long long g =
        (unsigned long long)(size_t)(W + (n0 + row) * D_ + k + col);
    unsigned int l = (unsigned int)(size_t)&Wlds[buf][row * 32 + col];
    ASYNC_LOAD_B128(l, g);
  };

  issue_w(0, 0);
  v8f acc[2][4];
#pragma unroll
  for (int i = 0; i < 2; ++i)
#pragma unroll
    for (int j = 0; j < 4; ++j) acc[i][j] = v8f{};

  for (int k = 0; k < D_; k += 32) {
    int buf = (k >> 5) & 1;
    if (k + 32 < D_) { issue_w(k + 32, buf ^ 1); WAIT_ASYNC(1); }
    else             { WAIT_ASYNC(0); }
    __syncthreads();
    v16bf a0 = load_frag(A, m0 + lm,      D_, k, lane);
    v16bf a1 = load_frag(A, m0 + 16 + lm, D_, k, lane);
    v16bf w0 = load_frag(&Wlds[buf][0],  0 + lm, 32, 0, lane);
    v16bf w1 = load_frag(&Wlds[buf][0], 16 + lm, 32, 0, lane);
    v16bf w2 = load_frag(&Wlds[buf][0], 32 + lm, 32, 0, lane);
    v16bf w3 = load_frag(&Wlds[buf][0], 48 + lm, 32, 0, lane);
    SCHED_FENCE();
    acc[0][0] = wmma_bf16(a0, w0, acc[0][0]);
    acc[0][1] = wmma_bf16(a0, w1, acc[0][1]);
    acc[0][2] = wmma_bf16(a0, w2, acc[0][2]);
    acc[0][3] = wmma_bf16(a0, w3, acc[0][3]);
    acc[1][0] = wmma_bf16(a1, w0, acc[1][0]);
    acc[1][1] = wmma_bf16(a1, w1, acc[1][1]);
    acc[1][2] = wmma_bf16(a1, w2, acc[1][2]);
    acc[1][3] = wmma_bf16(a1, w3, acc[1][3]);
    __syncthreads();
  }

#pragma unroll
  for (int i = 0; i < 2; ++i)
#pragma unroll
    for (int j = 0; j < 4; ++j)
#pragma unroll
      for (int r = 0; r < 8; ++r) {
        int m = m0 + i * 16 + r + half * 8;
        int n = n0 + j * 16 + lm;
        out[m * D_ + n] = acc[i][j][r] + bias[n];
      }
}

// -----------------------------------------------------------------------------
extern "C" void kernel_launch(void* const* d_in, const int* in_sizes, int n_in,
                              void* d_out, int out_size, void* d_ws, size_t ws_size,
                              hipStream_t stream) {
  (void)in_sizes; (void)n_in; (void)out_size; (void)ws_size;
  const float* x     = (const float*)d_in[0];
  const float* mask  = (const float*)d_in[1];
  const float* w_qkv = (const float*)d_in[2];
  const float* b_qkv = (const float*)d_in[3];
  const float* w_out = (const float*)d_in[4];
  const float* b_out = (const float*)d_in[5];
  float* out = (float*)d_out;

  char* ws = (char*)d_ws;
  size_t off = 0;
  auto carve = [&](size_t bytes) -> void* {
    void* p = ws + off;
    off += (bytes + 255) & ~(size_t)255;
    return p;
  };
  __bf16* xb    = (__bf16*)carve((size_t)B_ * S_ * D_ * 2);       // 8 MB
  __bf16* wqkvb = (__bf16*)carve((size_t)3 * D_ * D_ * 2);        // 6 MB
  __bf16* woutb = (__bf16*)carve((size_t)D_ * D_ * 2);            // 2 MB
  __bf16* Qb    = (__bf16*)carve((size_t)B_ * H_ * S_ * HD_ * 2); // 8 MB
  __bf16* Kb    = (__bf16*)carve((size_t)B_ * H_ * S_ * HD_ * 2); // 8 MB
  __bf16* Vt    = (__bf16*)carve((size_t)B_ * H_ * HD_ * S_ * 2); // 8 MB
  __bf16* vals  = (__bf16*)carve((size_t)B_ * S_ * D_ * 2);       // 8 MB

  cvt_bf16<<<512, 256, 0, stream>>>(x, xb, B_ * S_ * D_);
  cvt_bf16<<<512, 256, 0, stream>>>(w_qkv, wqkvb, 3 * D_ * D_);
  cvt_bf16<<<128, 256, 0, stream>>>(w_out, woutb, D_ * D_);

  // block tile 256x64, grid (N/64, M/256)
  qkv_gemm<<<dim3((3 * D_) / 64, (B_ * S_) / 256), 256, 0, stream>>>(
      xb, wqkvb, b_qkv, Qb, Kb, Vt);

  attn<<<dim3(B_ * H_, S_ / 64), 128, 0, stream>>>(Qb, Kb, Vt, mask, vals);

  out_gemm<<<dim3(D_ / 64, (B_ * S_) / 256), 256, 0, stream>>>(
      vals, woutb, b_out, out);
}
